// MPSLayer_18176301597298
// MI455X (gfx1250) — matrix operand order repaired and checked by reading, not compiled, not run on hardware
//
#include <hip/hip_runtime.h>
#include <hip/hip_bf16.h>

typedef __attribute__((ext_vector_type(16))) _Float16 v16h;
typedef __attribute__((ext_vector_type(8)))  _Float16 v8h;
typedef __attribute__((ext_vector_type(2)))  _Float16 v2h;
typedef __attribute__((ext_vector_type(8)))  float    v8f;

constexpr int B_   = 4096;
constexpr int S_   = 128;
constexpr int F_   = 8;
constexpr int D_   = 64;
constexpr int NSITE = S_ - 2;   // 126 middle sites
constexpr int KB_  = 16;        // K = 512 = 16 blocks of 32
constexpr int NT_  = 4;         // N = 64  = 4 tiles of 16
constexpr int VS2  = 68;        // v row stride in halves (34 dwords, conflict-free)
constexpr int PANEL_HALVES = KB_ * NT_ * 512;            // 32768 f16 = 64 KB
constexpr int PANEL_BYTES  = PANEL_HALVES * 2;           // 65536
constexpr int VBUF_HALVES  = 16 * VS2;                   // one 16x64(+pad) f16 tile
constexpr int SMEM_BYTES   = 2 * PANEL_BYTES + 2 * VBUF_HALVES * 2; // 135424

// permuted d-position: blocks of 4 stored as [4k, 4k+2, 4k+1, 4k+3]
__host__ __device__ inline int dperm(int d) {
    return (d & ~3) | ((d & 1) << 1) | ((d >> 1) & 1);
}

// ---- TDM availability / arity detection (device pass only) ----------------
#if defined(__has_builtin)
#  if __has_builtin(__builtin_amdgcn_tensor_load_to_lds)
#    define MPS_HAVE_TDM 1
#  endif
#endif
#ifndef MPS_HAVE_TDM
#  define MPS_HAVE_TDM 0
#endif
#if __has_include(<hip/amd_detail/amd_gfx1250_TDM.h>)
#  define MPS_TDM_6ARG 1   // therock-10.0 headers -> 6-arg builtin
#else
#  define MPS_TDM_6ARG 0   // ROCm 7.2 -> 5-arg builtin
#endif

#if MPS_HAVE_TDM
typedef unsigned int u32x4 __attribute__((ext_vector_type(4)));
typedef int          i32x8 __attribute__((ext_vector_type(8)));
typedef int          i32x4 __attribute__((ext_vector_type(4)));

// One-shot 64 KB contiguous DMA: global -> LDS, as a 1-D tensor of 8192
// 8-byte elements. D# layout per CDNA5 ISA 8.3/8.4.
__device__ __attribute__((always_inline))
inline void tdm_load_panel(const void* gsrc, unsigned lds_off) {
    unsigned long long ga = (unsigned long long)(size_t)gsrc;
    u32x4 g0;
    g0[0] = 1u;                                           // count=1 (valid), user mode
    g0[1] = lds_off;                                      // lds_addr
    g0[2] = (unsigned)(ga & 0xffffffffu);                 // global_addr[31:0]
    g0[3] = (unsigned)((ga >> 32) & 0x1ffffffu) | (2u << 30); // ga[56:32] | type=2
    i32x8 g1;
    g1[0] = (3 << 16);                                    // data_size=3 (8B); wg_mask=0
    g1[1] = (int)(8192u << 16);                           // tensor_dim0[15:0] @63:48
    g1[2] = (int)(1u << 16);                              // tensor_dim1 = 1
    g1[3] = (int)(8192u << 16);                           // tile_dim0 = 8192 @127:112
    g1[4] = 1;                                            // tile_dim1=1, tile_dim2=0
    g1[5] = 8192;                                         // tensor_dim0_stride[31:0]
    g1[6] = 0;
    g1[7] = 0;
    i32x4 gz = {0, 0, 0, 0};
#if MPS_TDM_6ARG
    i32x8 gz8 = {0, 0, 0, 0, 0, 0, 0, 0};
    __builtin_amdgcn_tensor_load_to_lds(g0, g1, gz, gz, gz8, 0);
#else
    __builtin_amdgcn_tensor_load_to_lds(g0, g1, gz, gz, 0);
#endif
}
#endif // MPS_HAVE_TDM

// ---------------------------------------------------------------------------
// Pre-pass: repack w_mid [126][64][8][64] fp32 -> f16 WMMA B-operand
// fragments; augmented K index k = d*8 + f so A[m,k] = v[m,d]*x[m,f].
// Fragment (site,kb,nt): lane = h*16+n holds, in half j, B[kb*32+16(j/8)+8h+j%8, nt*16+n].
// ---------------------------------------------------------------------------
__global__ void mps_pack_w_kernel(const float* __restrict__ w_mid,
                                  _Float16* __restrict__ wsW) {
    int tid = blockIdx.x * blockDim.x + threadIdx.x;
    int j    = tid & 15;
    int lane = (tid >> 4) & 31;
    int nt   = (tid >> 9) & 3;
    int kb   = (tid >> 11) & 15;
    int site = tid >> 15;
    if (site >= NSITE) return;
    int klocal = ((j >> 3) << 4) + ((lane >> 4) << 3) + (j & 7);
    int K = kb * 32 + klocal;
    int d = K >> 3, f = K & 7;
    int e = nt * 16 + (lane & 15);
    wsW[tid] = (_Float16)w_mid[((size_t)(site * D_ + d) * F_ + f) * D_ + e];
}

// ---------------------------------------------------------------------------
// Chain kernel: 4 waves / 16 batch rows; wave = e-tile. Per step:
//   v_new(16x64) = A(16x512) x B_s(512x64) via 16 wmma/wave, f32 accum.
// TDM double-buffers the 64KB B panel in LDS one site ahead; v is carried in
// LDS as f16 in permuted d-order so each A-build kb block needs one b32 load.
// ---------------------------------------------------------------------------
__global__ void __launch_bounds__(128)
mps_chain_kernel(const float* __restrict__ x,
                 const float* __restrict__ wfirst,
                 const float* __restrict__ wlast,
                 const _Float16* __restrict__ wsW,
                 float* __restrict__ out) {
    extern __shared__ char smem[];
    _Float16* lpanel = (_Float16*)smem;                       // 2 x 64KB panels
    _Float16* vb     = (_Float16*)(smem + 2 * PANEL_BYTES);   // 2 x 16 x VS2 f16

    const int tid  = threadIdx.x;
    const int wave = tid >> 5;          // 0..3 == e-tile (nt)
    const int lane = tid & 31;
    const int lo   = lane & 15;
    const int hi   = lane >> 4;
    const int nt   = wave;
    const int rowbase = blockIdx.x * 16;

#if MPS_HAVE_TDM
    // kick off panel for site 1 into LDS buffer 0 (overlaps with v0 compute)
    if (wave == 0)
        tdm_load_panel(wsW, (unsigned)(size_t)(lpanel));
#endif

    // ---- site 0: v0[m,d] = sum_f x[b,0,f] * w_first[f,d]  (stored permuted f16)
    {
        const int m     = tid & 15;
        const int dbase = (tid >> 4) * 8;
        const int bb    = rowbase + m;
        float x0[F_];
#pragma unroll
        for (int f = 0; f < F_; ++f) x0[f] = x[((size_t)bb * S_ + 0) * F_ + f];
#pragma unroll
        for (int d = 0; d < 8; ++d) {
            float acc = 0.f;
#pragma unroll
            for (int f = 0; f < F_; ++f) acc += x0[f] * wfirst[f * D_ + dbase + d];
            vb[m * VS2 + dperm(dbase + d)] = (_Float16)acc;
        }
    }
#if MPS_HAVE_TDM
    if (wave == 0) __builtin_amdgcn_s_wait_tensorcnt(0);
#endif
    __syncthreads();

    const int b = rowbase + lo;
    // permuted column position for this wave's accumulator scatter
    const int epos = dperm(nt * 16 + lo);

    // ---- middle sites
    for (int s = 1; s <= NSITE; ++s) {
        const _Float16* __restrict__ vold = vb + ((s - 1) & 1) * VBUF_HALVES;
        _Float16* __restrict__       vnew = vb + (s & 1) * VBUF_HALVES;

#if MPS_HAVE_TDM
        const _Float16* panel = lpanel + ((s - 1) & 1) * PANEL_HALVES;
        if (wave == 0 && s < NSITE)   // DMA next site's panel into other buffer
            tdm_load_panel(wsW + (size_t)s * PANEL_HALVES,
                           (unsigned)(size_t)(lpanel + (s & 1) * PANEL_HALVES));
#else
        const _Float16* panel = wsW + (size_t)(s - 1) * PANEL_HALVES;
        if (s < NSITE)
            __builtin_prefetch(wsW + (size_t)s * PANEL_HALVES + (size_t)tid * 256, 0, 1);
#endif

        _Float16 xh[F_];
#pragma unroll
        for (int f = 0; f < F_; ++f)
            xh[f] = (_Float16)x[((size_t)b * S_ + s) * F_ + f];

        v8f acc = {};
        for (int kb = 0; kb < KB_; ++kb) {
            // A fragment (16x32 f16): halves 0..7 -> d = kb*4+hi, f = j;
            //                         halves 8..15 -> d = kb*4+hi+2.
            // Permuted f16 storage puts both values in one dword.
            const v2h vp = *(const v2h*)(vold + lo * VS2 + kb * 4 + 2 * hi);
            const _Float16 hA = vp[0];
            const _Float16 hB = vp[1];
            v16h a;
#pragma unroll
            for (int f = 0; f < F_; ++f) {
                a[f]     = hA * xh[f];
                a[8 + f] = hB * xh[f];
            }
            const _Float16* fp = panel + ((size_t)(kb * NT_ + nt)) * 512 + lane * 16;
            v8h b0 = *(const v8h*)(fp);
            v8h b1 = *(const v8h*)(fp + 8);
            v16h bfrag = __builtin_shufflevector(b0, b1,
                0, 1, 2, 3, 4, 5, 6, 7, 8, 9, 10, 11, 12, 13, 14, 15);
            acc = __builtin_amdgcn_wmma_f32_16x16x32_f16(
                false, a, false, bfrag, (short)0, acc, false, false);
        }

        // scatter accumulator (C/D layout: VGPR j -> row hi*8+j, col = lo),
        // converting once to f16 at the permuted column position
#pragma unroll
        for (int j = 0; j < 8; ++j)
            vnew[(hi * 8 + j) * VS2 + epos] = (_Float16)acc[j];

#if MPS_HAVE_TDM
        if (wave == 0 && s < NSITE) __builtin_amdgcn_s_wait_tensorcnt(0);
#endif
        __syncthreads();   // publishes new v (and next panel) to all waves
    }

    // ---- last site: out[b] = sum_{d,f} v[b,d] * x[b,S-1,f] * w_last[d,f]
    if (wave == 0 && hi == 0) {
        const _Float16* vfin = vb + (NSITE & 1) * VBUF_HALVES;
        float xl[F_];
#pragma unroll
        for (int f = 0; f < F_; ++f)
            xl[f] = x[((size_t)b * S_ + (S_ - 1)) * F_ + f];
        float r = 0.f;
        for (int d = 0; d < D_; ++d) {
            float t = 0.f;
#pragma unroll
            for (int f = 0; f < F_; ++f) t += xl[f] * wlast[d * F_ + f];
            r += (float)vfin[lo * VS2 + dperm(d)] * t;
        }
        out[b] = r;
    }
}

extern "C" void kernel_launch(void* const* d_in, const int* in_sizes, int n_in,
                              void* d_out, int out_size, void* d_ws, size_t ws_size,
                              hipStream_t stream) {
    const float* x       = (const float*)d_in[0];
    const float* w_first = (const float*)d_in[1];
    const float* w_mid   = (const float*)d_in[2];
    const float* w_last  = (const float*)d_in[3];
    float*       out     = (float*)d_out;
    _Float16*    wsW     = (_Float16*)d_ws;   // 126 * 32768 halves ~= 8.3 MB

    const int total = NSITE * KB_ * NT_ * 32 * 16;   // 4,128,768 elements
    mps_pack_w_kernel<<<(total + 255) / 256, 256, 0, stream>>>(w_mid, wsW);
    mps_chain_kernel<<<B_ / 16, 128, SMEM_BYTES, stream>>>(x, w_first, w_last, wsW, out);
}
